// TSPNNet_GAT_34514357190743
// MI455X (gfx1250) — matrix operand order repaired and checked by reading, not compiled, not run on hardware
//
#include <hip/hip_runtime.h>
#include <hip/hip_bf16.h>
#include <math.h>

typedef __attribute__((ext_vector_type(16))) _Float16 v16h;
typedef __attribute__((ext_vector_type(8)))  _Float16 v8h;
typedef __attribute__((ext_vector_type(8)))  float    v8f;

static constexpr int Bb = 64;
static constexpr int Nn = 1024;
static constexpr int Cc = 128;
static constexpr int H1n = 256;
static constexpr int H2n = 128;
static constexpr int ACT = 1024;

// ---------------------------------------------------------------------------
// x = node_features @ W_in   ([B,N,3] @ [3,C] -> [B,N,C]) ; also write f16 copy
// ---------------------------------------------------------------------------
__global__ void k_in_proj(const float* __restrict__ nf,
                          const float* __restrict__ Win,
                          float* __restrict__ x, _Float16* __restrict__ xh) {
  int idx = blockIdx.x * blockDim.x + threadIdx.x;   // over B*N*C
  if (idx >= Bb * Nn * Cc) return;
  int c  = idx % Cc;
  int bn = idx / Cc;
  const float* f = nf + (size_t)bn * 3;
  float v = f[0] * Win[0 * Cc + c] + f[1] * Win[1 * Cc + c] + f[2] * Win[2 * Cc + c];
  x[idx]  = v;
  xh[idx] = (_Float16)v;
}

// ---------------------------------------------------------------------------
// ai = x @ a_src, aj = x @ a_dst   (per (b,n) dot over C)
// ---------------------------------------------------------------------------
__global__ void k_scores(const float* __restrict__ x,
                         const float* __restrict__ asrc,
                         const float* __restrict__ adst,
                         float* __restrict__ ai, float* __restrict__ aj) {
  int bn = blockIdx.x * blockDim.x + threadIdx.x;
  if (bn >= Bb * Nn) return;
  const float* row = x + (size_t)bn * Cc;
  float s1 = 0.f, s2 = 0.f;
  for (int c = 0; c < Cc; ++c) {
    float v = row[c];
    s1 += v * asrc[c];
    s2 += v * adst[c];
  }
  ai[bn] = s1;
  aj[bn] = s2;
}

__device__ __forceinline__ float leaky_mask(float e, int a) {
  e = e > 0.f ? e : 0.2f * e;
  return a ? e : -1.0e9f;
}

// ---------------------------------------------------------------------------
// Pass 1: per-row softmax statistics.  rowm = max, rinv = 1/sum(exp(e-m)).
// one 256-thread block per (b,i) row
// ---------------------------------------------------------------------------
__global__ void __launch_bounds__(256) k_softmax_stats(
    const int* __restrict__ adj, const float* __restrict__ ai,
    const float* __restrict__ aj, float* __restrict__ rowm,
    float* __restrict__ rinv) {
  int bi = blockIdx.x;            // b*N + i
  int b  = bi / Nn;
  const int*   arow = adj + (size_t)bi * Nn;
  const float* ajb  = aj + (size_t)b * Nn;
  float av = ai[bi];
  __shared__ float red[256];

  float m = -3.0e38f;
  for (int j = threadIdx.x; j < Nn; j += 256)
    m = fmaxf(m, leaky_mask(av + ajb[j], arow[j]));
  red[threadIdx.x] = m;
  __syncthreads();
  for (int s = 128; s > 0; s >>= 1) {
    if (threadIdx.x < s) red[threadIdx.x] = fmaxf(red[threadIdx.x], red[threadIdx.x + s]);
    __syncthreads();
  }
  m = red[0];
  __syncthreads();

  float sum = 0.f;
  for (int j = threadIdx.x; j < Nn; j += 256)
    sum += __expf(leaky_mask(av + ajb[j], arow[j]) - m);   // masked underflows to 0
  red[threadIdx.x] = sum;
  __syncthreads();
  for (int s = 128; s > 0; s >>= 1) {
    if (threadIdx.x < s) red[threadIdx.x] += red[threadIdx.x + s];
    __syncthreads();
  }
  if (threadIdx.x == 0) { rowm[bi] = m; rinv[bi] = 1.0f / red[0]; }
}

// ---------------------------------------------------------------------------
// Pass 2: out[b, i0:i0+16, :] = relu( (exp(e-m)*rinv) @ x )  via WMMA f16.
// 256 threads = 8 waves; wave w owns 16 output columns. 64 j's per LDS stage,
// 2 WMMA per stage.  Also writes f16 copy for the next layer.
// ---------------------------------------------------------------------------
__global__ void __launch_bounds__(256) k_gat_agg(
    const int* __restrict__ adj, const float* __restrict__ ai,
    const float* __restrict__ aj, const float* __restrict__ rowm,
    const float* __restrict__ rinv, const _Float16* __restrict__ xh,
    float* __restrict__ out, _Float16* __restrict__ xh_out) {
  int b     = blockIdx.x / (Nn / 16);
  int itile = blockIdx.x % (Nn / 16);
  int i0    = itile * 16;

  __shared__ _Float16 As[16][64];        // p tile, row-major [i_local][j_local]
  __shared__ _Float16 Bs_t[Cc][72];      // x tile, k-major per column (pad 8)

  int t    = threadIdx.x;
  int lane = t & 31;
  int wave = t >> 5;                     // 0..7
  int c0   = wave * 16;

  const float* aib  = ai   + (size_t)b * Nn;
  const float* ajb  = aj   + (size_t)b * Nn;
  const float* mrow = rowm + (size_t)b * Nn;

  // A-fill coordinates: 4 consecutive j per thread
  const int fil = (4 * t) >> 6;          // i_local (0..15)
  const int fjl = (4 * t) & 63;          // j_local base
  const int   fi   = i0 + fil;
  const float f_av = aib[fi];
  const float f_mi = mrow[fi];
  const int* arow_base = adj + ((size_t)b * Nn + fi) * (size_t)Nn;

  v8f acc = {0.f, 0.f, 0.f, 0.f, 0.f, 0.f, 0.f, 0.f};

  for (int j0 = 0; j0 < Nn; j0 += 64) {
    // ---- fill As: 1024 entries, 4 consecutive-j per thread (coalesced adj)
    {
      const int* arow = arow_base + j0;
#pragma unroll
      for (int q = 0; q < 4; ++q) {
        int j   = fjl + q;
        float e = leaky_mask(f_av + ajb[j0 + j], arow[j]);
        As[fil][j] = (_Float16)__expf(e - f_mi);
      }
      if (j0 + 64 < Nn) __builtin_prefetch(arow + 64 + fjl, 0, 0);
    }
    // ---- fill Bs_t (transpose stage): 64x128 halves = 4096 dwords, 16/thread
    {
      const uint32_t* src = (const uint32_t*)(xh + ((size_t)b * Nn + j0) * Cc);
#pragma unroll
      for (int q = 0; q < 16; ++q) {
        int dw = t + 256 * q;            // dword index over tile
        int j  = dw >> 6;                // row (j_local)
        int c  = (dw & 63) * 2;          // column pair
        union { uint32_t u; _Float16 f[2]; } w;
        w.u = src[dw];
        Bs_t[c][j]     = w.f[0];
        Bs_t[c + 1][j] = w.f[1];
      }
      if (j0 + 64 < Nn) __builtin_prefetch(src + 4096 + t, 0, 0);
    }
    __syncthreads();

    // ---- two K=32 WMMA steps per stage
    int M  = lane & 15;
    int h  = lane >> 4;
    int nl = lane & 15;
#pragma unroll
    for (int u = 0; u < 2; ++u) {
      // A frag: lo = K(u*32 + h*8 + 0..7), hi = K(u*32 + 16 + h*8 + 0..7)
      v8h alo = *(const v8h*)&As[M][u * 32 + h * 8];
      v8h ahi = *(const v8h*)&As[M][u * 32 + 16 + h * 8];
      v16h afrag = __builtin_shufflevector(alo, ahi, 0, 1, 2, 3, 4, 5, 6, 7,
                                           8, 9, 10, 11, 12, 13, 14, 15);
      // B frag: lo = K(u*32 + h*8 + 0..7), hi = K(u*32 + 16 + h*8 + 0..7) at column c0+nl
      v8h blo = *(const v8h*)&Bs_t[c0 + nl][u * 32 + h * 8];
      v8h bhi = *(const v8h*)&Bs_t[c0 + nl][u * 32 + 16 + h * 8];
      v16h bfrag = __builtin_shufflevector(blo, bhi, 0, 1, 2, 3, 4, 5, 6, 7,
                                           8, 9, 10, 11, 12, 13, 14, 15);
      acc = __builtin_amdgcn_wmma_f32_16x16x32_f16(
          false, afrag, false, bfrag, (short)0, acc, false, false);
    }
    __syncthreads();
  }

  // ---- epilogue: multiply by 1/rowsum, relu, store f32 + f16
  int h = lane >> 4, nl = lane & 15;
#pragma unroll
  for (int r = 0; r < 8; ++r) {
    int i = i0 + r + 8 * h;              // C/D layout: M = r + 8*(lane/16)
    float v = acc[r] * rinv[(size_t)b * Nn + i];
    v = v > 0.f ? v : 0.f;
    size_t o = ((size_t)b * Nn + i) * Cc + c0 + nl;
    out[o]    = v;
    xh_out[o] = (_Float16)v;
  }
}

// ---------------------------------------------------------------------------
// mean over nodes: pooled[b,c] = mean_n x[b,n,c]
// ---------------------------------------------------------------------------
__global__ void k_mean(const float* __restrict__ x, float* __restrict__ pooled) {
  int idx = blockIdx.x * blockDim.x + threadIdx.x;   // B*C
  if (idx >= Bb * Cc) return;
  int b = idx / Cc, c = idx % Cc;
  float s = 0.f;
  for (int n = 0; n < Nn; ++n) s += x[((size_t)b * Nn + n) * Cc + c];
  pooled[idx] = s * (1.0f / Nn);
}

// ---------------------------------------------------------------------------
// generic small dense: out[m,n] = act(in[m,:] @ W[:,n] + bias[n])
// ---------------------------------------------------------------------------
__global__ void k_dense(const float* __restrict__ in, const float* __restrict__ W,
                        const float* __restrict__ bias, float* __restrict__ out,
                        int M, int K, int Nout, int relu) {
  int idx = blockIdx.x * blockDim.x + threadIdx.x;
  if (idx >= M * Nout) return;
  int m = idx / Nout, n = idx % Nout;
  float s = bias[n];
  for (int k = 0; k < K; ++k) s += in[(size_t)m * K + k] * W[(size_t)k * Nout + n];
  if (relu) s = fmaxf(s, 0.f);
  out[idx] = s;
}

// ---------------------------------------------------------------------------
// row softmax over ACTION logits -> pi  (keep accurate expf for final output)
// ---------------------------------------------------------------------------
__global__ void __launch_bounds__(256) k_row_softmax(const float* __restrict__ logits,
                                                     float* __restrict__ out) {
  int b = blockIdx.x;
  const float* row = logits + (size_t)b * ACT;
  __shared__ float red[256];
  float m = -3.0e38f;
  for (int j = threadIdx.x; j < ACT; j += 256) m = fmaxf(m, row[j]);
  red[threadIdx.x] = m;
  __syncthreads();
  for (int s = 128; s > 0; s >>= 1) {
    if (threadIdx.x < s) red[threadIdx.x] = fmaxf(red[threadIdx.x], red[threadIdx.x + s]);
    __syncthreads();
  }
  m = red[0];
  __syncthreads();
  float sum = 0.f;
  for (int j = threadIdx.x; j < ACT; j += 256) sum += expf(row[j] - m);
  red[threadIdx.x] = sum;
  __syncthreads();
  for (int s = 128; s > 0; s >>= 1) {
    if (threadIdx.x < s) red[threadIdx.x] += red[threadIdx.x + s];
    __syncthreads();
  }
  float inv = 1.0f / red[0];
  for (int j = threadIdx.x; j < ACT; j += 256)
    out[(size_t)b * ACT + j] = expf(row[j] - m) * inv;
}

// ---------------------------------------------------------------------------
extern "C" void kernel_launch(void* const* d_in, const int* in_sizes, int n_in,
                              void* d_out, int out_size, void* d_ws, size_t ws_size,
                              hipStream_t stream) {
  (void)in_sizes; (void)n_in; (void)out_size; (void)ws_size;
  const float* nf    = (const float*)d_in[0];
  const int*   adj   = (const int*)d_in[1];
  const float* Win   = (const float*)d_in[2];
  const float* asrc1 = (const float*)d_in[3];
  const float* adst1 = (const float*)d_in[4];
  const float* asrc2 = (const float*)d_in[5];
  const float* adst2 = (const float*)d_in[6];
  const float* W1    = (const float*)d_in[7];
  const float* b1    = (const float*)d_in[8];
  const float* W2    = (const float*)d_in[9];
  const float* b2    = (const float*)d_in[10];
  const float* Wpi   = (const float*)d_in[11];
  const float* bpi   = (const float*)d_in[12];
  const float* Wv    = (const float*)d_in[13];
  const float* bv    = (const float*)d_in[14];
  float* out = (float*)d_out;

  const size_t BNC = (size_t)Bb * Nn * Cc;
  char* ws = (char*)d_ws;
  size_t off = 0;
  float*    xA     = (float*)(ws + off); off += BNC * 4;
  float*    xB     = (float*)(ws + off); off += BNC * 4;
  _Float16* xhA    = (_Float16*)(ws + off); off += BNC * 2;
  _Float16* xhB    = (_Float16*)(ws + off); off += BNC * 2;
  float*    ai     = (float*)(ws + off); off += (size_t)Bb * Nn * 4;
  float*    aj     = (float*)(ws + off); off += (size_t)Bb * Nn * 4;
  float*    rowm   = (float*)(ws + off); off += (size_t)Bb * Nn * 4;
  float*    rinv   = (float*)(ws + off); off += (size_t)Bb * Nn * 4;
  float*    pooled = (float*)(ws + off); off += (size_t)Bb * Cc * 4;
  float*    h1     = (float*)(ws + off); off += (size_t)Bb * H1n * 4;
  float*    h2     = (float*)(ws + off); off += (size_t)Bb * H2n * 4;
  float*    logits = (float*)(ws + off); off += (size_t)Bb * ACT * 4;

  // input projection (writes f32 + f16)
  k_in_proj<<<(int)((BNC + 255) / 256), 256, 0, stream>>>(nf, Win, xA, xhA);

  const float* asrc[2] = {asrc1, asrc2};
  const float* adst[2] = {adst1, adst2};
  float*    xin = xA;   float*    xout = xB;
  _Float16* hin = xhA;  _Float16* hout = xhB;
  for (int l = 0; l < 2; ++l) {
    k_scores<<<(Bb * Nn + 255) / 256, 256, 0, stream>>>(xin, asrc[l], adst[l], ai, aj);
    k_softmax_stats<<<Bb * Nn, 256, 0, stream>>>(adj, ai, aj, rowm, rinv);
    k_gat_agg<<<Bb * (Nn / 16), 256, 0, stream>>>(adj, ai, aj, rowm, rinv, hin, xout, hout);
    float* tf = xin; xin = xout; xout = tf;
    _Float16* th = hin; hin = hout; hout = th;
  }
  // xin now holds layer-2 output (f32)
  k_mean<<<(Bb * Cc + 255) / 256, 256, 0, stream>>>(xin, pooled);
  k_dense<<<(Bb * H1n + 255) / 256, 256, 0, stream>>>(pooled, W1, b1, h1, Bb, Cc, H1n, 1);
  k_dense<<<(Bb * H2n + 255) / 256, 256, 0, stream>>>(h1, W2, b2, h2, Bb, H1n, H2n, 1);
  k_dense<<<(Bb * ACT + 255) / 256, 256, 0, stream>>>(h2, Wpi, bpi, logits, Bb, H2n, ACT, 0);
  k_dense<<<(Bb + 255) / 256, 256, 0, stream>>>(h2, Wv, bv, out + (size_t)Bb * ACT, Bb, H2n, 1, 0);
  k_row_softmax<<<Bb, 256, 0, stream>>>(logits, out);
}
